// CoIL_37855841747602
// MI455X (gfx1250) — matrix operand order repaired and checked by reading, not compiled, not run on hardware
//
#include <hip/hip_runtime.h>

// ---------------------------------------------------------------------------
// MoE-style MLP:  out[b,:] = (relu(x[b,:] @ W1^T + b1) @ Wb[u[b]]^T) + bb[u[b]]
// B=1e6, IN=64, H=128, OUT=2, 3 heads.
//
// Roofline: x stream = 256 MB @ 23.3 TB/s => ~11.5 us floor. All GEMMs via
// v_wmma_f32_16x16x32_f16 (f16 in, f32 acc) so the 16.9 GFLOP hides under it.
//
// Per-wave 16-row batch tile, grid-stride over 62500 tiles:
//   stage 1:  H^T = W1(128x64) @ x^T(64x16): 16 WMMA in a double-buffered
//             pipeline (next A tile's ds_load_b128 pair issued before each
//             WMMA). W1(f16) lives in LDS (16 KB / workgroup); bias lives in
//             64 loop-invariant VGPRs and enters as the C operand of the
//             first K-chunk (bias is free).
//   stage 2:  Y^T = WbAll(6x128 pad 16) @ H^T(128x16): 4 WMMA on two
//             independent accumulator chains. The stage-2 K dimension is
//             permuted so its B operand is built IN-LANE from the stage-1
//             C-layout accumulators (relu+cvt only, zero shuffles); the
//             permutation is baked into the once-per-wave WbAll A preload.
//   select:   per-row head pick (u in {0,1,2}) via cndmask, one b64 store.
// ---------------------------------------------------------------------------

typedef __attribute__((ext_vector_type(16))) _Float16 v16h;
typedef __attribute__((ext_vector_type(8)))  _Float16 v8h;
typedef __attribute__((ext_vector_type(8)))  float    v8f;
typedef __attribute__((ext_vector_type(4)))  float    v4f;
typedef __attribute__((ext_vector_type(2)))  float    v2f;

#define B_ROWS  1000000
#define IN_DIM  64
#define H_DIM   128
#define NBLK    (B_ROWS / 16)

union U16h { v16h v; v8h h[2]; };

__global__ __launch_bounds__(256) void CoIL_moe_mlp_wmma(
    const float* __restrict__ x,   // [B, 64]
    const int*   __restrict__ u,   // [B]
    const float* __restrict__ W1,  // [128, 64]
    const float* __restrict__ b1,  // [128]
    const float* __restrict__ Wb,  // [3, 2, 128]
    const float* __restrict__ bb,  // [3, 2]
    float*       __restrict__ out, // [B, 2]
    int nwaves)
{
  __shared__ _Float16 sW1[H_DIM * IN_DIM]; // 16 KB, row-major f16

  const int lane = threadIdx.x & 31;
  const int wave = blockIdx.x * (blockDim.x >> 5) + (threadIdx.x >> 5);
  const int m    = lane & 15;   // A-layout row / C-layout column
  const int hi   = lane >> 4;   // lane half

  // ---- cooperative LDS fill (once per workgroup) ---------------------------
  for (int i = threadIdx.x; i < H_DIM * IN_DIM; i += blockDim.x)
    sW1[i] = (_Float16)W1[i];
  __syncthreads();

  // ---- once per wave: bias in stage-1 C layout, resident in VGPRs ----------
  // acc C layout: reg r, lane half hi -> h = 16*t + r + 8*hi  (contig in r)
  v8f biasv[8];
#pragma unroll
  for (int t = 0; t < 8; ++t) {
    const float* p = b1 + 16 * t + 8 * hi;
    v4f f0 = *(const v4f*)(p);
    v4f f1 = *(const v4f*)(p + 4);
#pragma unroll
    for (int j = 0; j < 4; ++j) { biasv[t][j] = f0[j]; biasv[t][4 + j] = f1[j]; }
  }

  // ---- once per wave: WbAll (6x128 padded to 16) A-layout, K permuted ------
  // K-permutation: kappa = 32*(t>>1) + 16*hi + 8*(t&1) + r  (bijective), so
  // stage-2 B chunk q = acc[2q][0..7] ++ acc[2q+1][0..7] entirely in-lane.
  v16h aWb[4];
#pragma unroll
  for (int q = 0; q < 4; ++q) {
    v16h a;
#pragma unroll
    for (int j = 0; j < 16; ++j) {
      int Kl  = 16 * hi + j;
      int kap = 32 * q + Kl;
      int h   = 32 * (kap >> 5) + 16 * ((kap >> 3) & 1) + (kap & 7)
              + 8 * ((kap >> 4) & 1);
      float w = (m < 6) ? Wb[m * H_DIM + h] : 0.0f;  // row m = head*2 + o
      a[j] = (_Float16)w;
    }
    aWb[q] = a;
  }

  // ---- stream the batch: 16 rows per wave-iteration ------------------------
  for (int blk = wave; blk < NBLK; blk += nwaves) {
    const int rowBase = blk * 16;
    const float* xr = x + (rowBase + m) * IN_DIM + 16 * hi;

    // next sweep's rows into L2 while we compute
    __builtin_prefetch(xr + (size_t)nwaves * 16 * IN_DIM, 0, 0);

    // opaque zero: keeps the loop-invariant LDS reads per-iteration so W1
    // stays in LDS (not hoisted into VGPRs)
    int opq = 0;
    asm volatile("" : "+v"(opq));
    // per-lane A base: row m of each 16-row W1 tile, K offset 16*hi
    const _Float16* swp = sW1 + opq + m * IN_DIM + 16 * hi;

    // B operand (x^T): lane half hi holds K = 16*hi + j  -> 16 contig floats
    v16h bx[2];
#pragma unroll
    for (int c = 0; c < 2; ++c) {
      v4f f0 = *(const v4f*)(xr + 32 * c);
      v4f f1 = *(const v4f*)(xr + 32 * c + 4);
      v4f f2 = *(const v4f*)(xr + 32 * c + 8);
      v4f f3 = *(const v4f*)(xr + 32 * c + 12);
      v16h b;
#pragma unroll
      for (int j = 0; j < 4; ++j) {
        b[j]      = (_Float16)f0[j];
        b[4 + j]  = (_Float16)f1[j];
        b[8 + j]  = (_Float16)f2[j];
        b[12 + j] = (_Float16)f3[j];
      }
      bx[c] = b;
    }

    // stage 1: 16-step double-buffered pipeline over (chunk c, tile t);
    // the (s+1) A operand is loaded from LDS before the WMMA of step s.
    v8f acc[8];
    U16h ab[2];
    ab[0].h[0] = *(const v8h*)(swp);
    ab[0].h[1] = *(const v8h*)(swp + 8);
#pragma unroll
    for (int s = 0; s < 16; ++s) {
      const int c = s >> 3;
      const int t = s & 7;
      if (s < 15) {
        const int c2 = (s + 1) >> 3;
        const int t2 = (s + 1) & 7;
        const _Float16* pw = swp + t2 * (16 * IN_DIM) + c2 * 32;
        ab[(s + 1) & 1].h[0] = *(const v8h*)(pw);
        ab[(s + 1) & 1].h[1] = *(const v8h*)(pw + 8);
      }
      acc[t] = __builtin_amdgcn_wmma_f32_16x16x32_f16(
          false, ab[s & 1].v, false, bx[c], (short)0,
          (c == 0) ? biasv[t] : acc[t], false, false);
    }

    // stage 2: per-q relu+pack (in-lane, permuted K) then WMMA immediately on
    // alternating accumulator chains (q reads tiles 2q,2q+1: >=2 WMMAs back)
    v8f y0 = {}, y1 = {};
#pragma unroll
    for (int q = 0; q < 4; ++q) {
      v16h b;
#pragma unroll
      for (int j = 0; j < 8; ++j) {
        b[j]     = (_Float16)fmaxf(acc[2 * q][j],     0.0f);
        b[8 + j] = (_Float16)fmaxf(acc[2 * q + 1][j], 0.0f);
      }
      if (q & 1)
        y1 = __builtin_amdgcn_wmma_f32_16x16x32_f16(
            false, aWb[q], false, b, (short)0, y1, false, false);
      else
        y0 = __builtin_amdgcn_wmma_f32_16x16x32_f16(
            false, aWb[q], false, b, (short)0, y0, false, false);
    }
    v8f y = y0 + y1;

    // per-row head selection + store (lanes 0..15 hold M = 0..5 in regs 0..5)
    if (lane < 16) {
      const int row = rowBase + m;
      const int kk  = u[row];
      float o0 = (kk == 0) ? y[0] : ((kk == 1) ? y[2] : y[4]);
      float o1 = (kk == 0) ? y[1] : ((kk == 1) ? y[3] : y[5]);
      o0 += bb[kk * 2 + 0];
      o1 += bb[kk * 2 + 1];
      v2f o; o[0] = o0; o[1] = o1;
      *(v2f*)(out + (size_t)row * 2) = o;
    }
  }
}

extern "C" void kernel_launch(void* const* d_in, const int* in_sizes, int n_in,
                              void* d_out, int out_size, void* d_ws, size_t ws_size,
                              hipStream_t stream) {
  const float* x  = (const float*)d_in[0];
  const int*   u  = (const int*)  d_in[1];
  const float* W1 = (const float*)d_in[2];
  const float* b1 = (const float*)d_in[3];
  const float* Wb = (const float*)d_in[4];
  const float* bb = (const float*)d_in[5];
  float* out = (float*)d_out;
  (void)in_sizes; (void)n_in; (void)out_size; (void)d_ws; (void)ws_size;

  // 256 threads = 8 wave32 per WG share the LDS weights; 256 WGs -> 2048
  // independent waves, ~30 batch tiles each (LDS fill amortized ~30x).
  dim3 grid(256), block(256);
  const int nwaves = 256 * (256 / 32);
  CoIL_moe_mlp_wmma<<<grid, block, 0, stream>>>(x, u, W1, b1, Wb, bb, out, nwaves);
}
